// H2VQ_bodyhand_23287312679012
// MI455X (gfx1250) — compile-verified, matrix operand order checked by reading, not run on hardware
//
#include <hip/hip_runtime.h>
#include <stdint.h>

// Problem constants (from the reference)
#define NB_CODE   512
#define CODE_DIM  512
#define NBATCH    32
#define TLEN      2048
#define M_TOTAL   (NBATCH * TLEN)     // 65536 vectors
#define KTILES    (NB_CODE / 16)      // 32
#define DCHUNKS   (CODE_DIM / 32)     // 16

typedef __attribute__((ext_vector_type(16))) __bf16  v16bf;
typedef __attribute__((ext_vector_type(8)))  float   v8f;
typedef __attribute__((ext_vector_type(4)))  unsigned int u32x4;
typedef __attribute__((address_space(3)))    unsigned char* lds_p;

union Frag {
    u32x4        q[2];
    unsigned int u[8];
    v16bf        v;
};

// round-to-nearest-even f32 -> bf16 bits
__device__ __forceinline__ unsigned int f2bf_bits(float f) {
    unsigned int u = __float_as_uint(f);
    return (u + 0x7FFFu + ((u >> 16) & 1u)) >> 16;
}
__device__ __forceinline__ float bfhi_to_f(unsigned int b) {
    return __uint_as_float(b << 16);
}
// ISA 16-bit A/B fragment layout: lane<16 holds K {0..7,16..23}, lane>=16 holds {8..15,24..31};
// dword r holds the K-pair starting at this offset:
__device__ __forceinline__ int doff(int lane, int r) {
    return ((r & 4) << 2) + ((lane & 16) >> 1) + ((r & 3) << 1);
}

// CDNA5 async memory->LDS copy: 16B per lane per instruction, tracked by ASYNCcnt.
// vdst = per-lane LDS byte address, vaddr = per-lane 64-bit global address.
__device__ __forceinline__ void async_copy64B(lds_p lds_dst, const unsigned int* gsrc) {
    asm volatile("global_load_async_to_lds_b128 %0, %1, off\n\t"
                 "global_load_async_to_lds_b128 %0, %1, off offset:16\n\t"
                 "global_load_async_to_lds_b128 %0, %1, off offset:32\n\t"
                 "global_load_async_to_lds_b128 %0, %1, off offset:48"
                 :: "v"(lds_dst), "v"(gsrc) : "memory");
}
__device__ __forceinline__ void wait_async0() {
    asm volatile("s_wait_asynccnt 0x0" ::: "memory");
}

// ---------------- prep: zero accumulators ----------------
__global__ void vq_zero(float* counts, float* loss) {
    int i = blockIdx.x * blockDim.x + threadIdx.x;
    if (i < NB_CODE) counts[i] = 0.0f;
    if (i == 0) loss[0] = 0.0f;
}

// ---------------- prep: codebook squared norms ----------------
__global__ void vq_prep_csq(const float* __restrict__ cb, float* __restrict__ c_sq) {
    __shared__ float red[256];
    int k = blockIdx.x;
    float s = 0.0f;
    for (int d = threadIdx.x; d < CODE_DIM; d += 256) {
        float v = cb[(size_t)k * CODE_DIM + d];
        s += v * v;
    }
    red[threadIdx.x] = s;
    __syncthreads();
    for (int off = 128; off > 0; off >>= 1) {
        if ((int)threadIdx.x < off) red[threadIdx.x] += red[threadIdx.x + off];
        __syncthreads();
    }
    if (threadIdx.x == 0) c_sq[k] = red[0];
}

// ---------------- prep: swizzle codebook into WMMA B fragments (bf16 hi/lo) ----------------
// Layout: bswz[((kt*16 + dc)*32 + lane)*8 + r], each dword = packed bf16 pair (d, d+1)
__global__ void vq_prep_swizzle(const float* __restrict__ cb,
                                unsigned int* __restrict__ bhi,
                                unsigned int* __restrict__ blo) {
    int blk  = blockIdx.x;          // kt*16 + dc  (512 blocks)
    int kt   = blk >> 4;
    int dc   = blk & 15;
    int lane = threadIdx.x >> 3;
    int r    = threadIdx.x & 7;
    int k    = kt * 16 + (lane & 15);
    int d    = dc * 32 + doff(lane, r);

    float a = cb[(size_t)k * CODE_DIM + d];
    float b = cb[(size_t)k * CODE_DIM + d + 1];
    unsigned int ah = f2bf_bits(a), bh = f2bf_bits(b);
    unsigned int al = f2bf_bits(a - bfhi_to_f(ah));
    unsigned int bl = f2bf_bits(b - bfhi_to_f(bh));

    size_t idx = (size_t)blk * 256 + threadIdx.x;  // == ((kt*16+dc)*32+lane)*8 + r
    bhi[idx] = ah | (bh << 16);
    blo[idx] = al | (bl << 16);
}

// ---------------- main: distances via bf16x3 WMMA, argmin, gather, loss, counts ----------------
__global__ __launch_bounds__(256)
void vq_main(const float* __restrict__ x, const float* __restrict__ cb,
             const unsigned int* __restrict__ bhi, const unsigned int* __restrict__ blo,
             const float* __restrict__ c_sq,
             float* __restrict__ out_x, float* __restrict__ out_idx,
             float* __restrict__ counts, float* __restrict__ loss) {

    // double-buffered B tile: per buffer, hi frags at dwords [0,4096), lo at [4096,8192)
    __shared__ unsigned int ldsB[2][8192];   // 2 x 32 KB = 64 KB

    const int tid  = threadIdx.x;
    const int wave = tid >> 5;
    const int lane = tid & 31;
    const int row  = lane & 15;
    const int mt   = blockIdx.x * 8 + wave;  // 4096 m-tiles total
    const int m0   = mt * 16;
    const int n    = m0 >> 11;               // / TLEN
    const int t0   = m0 & 2047;              // % TLEN

    const float* xbase = x + ((size_t)n * CODE_DIM) * TLEN + t0 + row;  // + d*TLEN

    // ---- stage the 16x512 x tile as bf16 hi/lo WMMA A-fragments, kept in VGPRs ----
    Frag ahi[DCHUNKS], alo[DCHUNKS];
#pragma unroll
    for (int dc = 0; dc < DCHUNKS; ++dc) {
#pragma unroll
        for (int r = 0; r < 8; ++r) {
            int d   = dc * 32 + doff(lane, r);
            float a = xbase[(size_t)d * TLEN];
            float b = xbase[(size_t)(d + 1) * TLEN];
            unsigned int h0 = f2bf_bits(a), h1 = f2bf_bits(b);
            unsigned int l0 = f2bf_bits(a - bfhi_to_f(h0));
            unsigned int l1 = f2bf_bits(b - bfhi_to_f(h1));
            ahi[dc].u[r] = h0 | (h1 << 16);
            alo[dc].u[r] = l0 | (l1 << 16);
        }
    }

    float bestD[8];
    int   bestI[8];
#pragma unroll
    for (int r = 0; r < 8; ++r) { bestD[r] = 3.4e38f; bestI[r] = 0; }

    // per-thread async-copy cursors: thread tid owns dwords [tid*16, tid*16+16) of each region
    lds_p lbuf0 = (lds_p)&ldsB[0][0] + (size_t)tid * 64;
    lds_p lbuf1 = (lds_p)&ldsB[1][0] + (size_t)tid * 64;

    {   // prologue: async-stage k-tile 0 into buffer 0
        async_copy64B(lbuf0,         bhi + (size_t)tid * 16);
        async_copy64B(lbuf0 + 16384, blo + (size_t)tid * 16);
        wait_async0();
    }
    __syncthreads();

    for (int kt = 0; kt < KTILES; ++kt) {
        const bool more = (kt + 1 < KTILES);
        if (more) {  // async-stage next tile into the other buffer; drains behind the WMMAs
            lds_p dst = ((kt + 1) & 1) ? lbuf1 : lbuf0;
            const size_t g = (size_t)(kt + 1) * 4096 + (size_t)tid * 16;
            async_copy64B(dst,         bhi + g);
            async_copy64B(dst + 16384, blo + g);
        }

        const unsigned int* cur = &ldsB[kt & 1][0];

        // three independent accumulator chains -> consecutive WMMAs are independent
        v8f aHH = {0.f,0.f,0.f,0.f,0.f,0.f,0.f,0.f};
        v8f aLH = {0.f,0.f,0.f,0.f,0.f,0.f,0.f,0.f};
        v8f aHL = {0.f,0.f,0.f,0.f,0.f,0.f,0.f,0.f};
#pragma unroll
        for (int dc = 0; dc < DCHUNKS; ++dc) {
            Frag fbh, fbl;
            const u32x4* ph = (const u32x4*)(cur + (dc * 32 + lane) * 8);
            const u32x4* pl = (const u32x4*)(cur + 4096 + (dc * 32 + lane) * 8);
            fbh.q[0] = ph[0]; fbh.q[1] = ph[1];
            fbl.q[0] = pl[0]; fbl.q[1] = pl[1];
            // split-precision fp32 emulation: hi*hi + lo*hi + hi*lo
            aHH = __builtin_amdgcn_wmma_f32_16x16x32_bf16(false, ahi[dc].v, false, fbh.v,
                                                          (short)0, aHH, false, false);
            aLH = __builtin_amdgcn_wmma_f32_16x16x32_bf16(false, alo[dc].v, false, fbh.v,
                                                          (short)0, aLH, false, false);
            aHL = __builtin_amdgcn_wmma_f32_16x16x32_bf16(false, ahi[dc].v, false, fbl.v,
                                                          (short)0, aHL, false, false);
        }

        // dist[m][k] = ||c||^2 - 2 x.c   (||x||^2 is argmin-invariant)
        float csq  = c_sq[kt * 16 + row];
        int   kidx = kt * 16 + row;
#pragma unroll
        for (int r = 0; r < 8; ++r) {
            float dist = csq - 2.0f * (aHH[r] + aLH[r] + aHL[r]);
            if (dist < bestD[r]) { bestD[r] = dist; bestI[r] = kidx; }
        }

        if (more) {
            wait_async0();     // my async writes for the next tile have landed in LDS
            __syncthreads();   // everyone's writes visible; everyone done reading this tile
        }
    }

    // cross-lane argmin within each 16-lane half (rows 0-7 live in lanes 0-15, rows 8-15 in 16-31)
#pragma unroll
    for (int off = 1; off < 16; off <<= 1) {
#pragma unroll
        for (int r = 0; r < 8; ++r) {
            float od = __shfl_xor(bestD[r], off, 16);
            int   oi = __shfl_xor(bestI[r], off, 16);
            if (od < bestD[r] || (od == bestD[r] && oi < bestI[r])) {
                bestD[r] = od; bestI[r] = oi;
            }
        }
    }

    // idx-exchange scratch aliases buffer 0 (dead: last k-tile was read from buffer 1)
    int* idxShared = (int*)&ldsB[0][0];
    if (lane == 0 || lane == 16) {
        int rbase = (lane >> 4) * 8;
#pragma unroll
        for (int r = 0; r < 8; ++r) {
            int mrow = rbase + r;
            idxShared[wave * 16 + mrow] = bestI[r];
            out_idx[m0 + mrow] = (float)bestI[r];
            atomicAdd(counts + bestI[r], 1.0f);
        }
    }
    __syncthreads();

    // ---- dequantize/gather epilogue + exact fp32 commit-loss accumulation ----
    const int    half   = lane >> 4;  // even/odd d
    const int    myCode = idxShared[wave * 16 + row];
    const float* crow   = cb + (size_t)myCode * CODE_DIM;
    float*       obase  = out_x + ((size_t)n * CODE_DIM) * TLEN + t0 + row;
    float lsum = 0.0f;
    for (int d0 = 0; d0 < CODE_DIM; d0 += 2) {
        int   d  = d0 + half;
        float c  = crow[d];
        float xv = xbase[(size_t)d * TLEN];
        obase[(size_t)d * TLEN] = c;       // straight-through forward == codebook row
        float df = xv - c;
        lsum += df * df;
    }
#pragma unroll
    for (int off = 16; off > 0; off >>= 1) lsum += __shfl_xor(lsum, off, 32);
    if (lane == 0) atomicAdd(loss, lsum);
}

// ---------------- finalize: perplexity + commit loss scalars ----------------
__global__ void vq_final(const float* __restrict__ counts, const float* __restrict__ loss,
                         float* __restrict__ out_scalars) {
    __shared__ float red[256];
    int tid = threadIdx.x;
    float s = 0.0f;
    for (int k = tid; k < NB_CODE; k += 256) {
        float p = counts[k] * (1.0f / (float)M_TOTAL);
        s += p * logf(p + 1e-7f);
    }
    red[tid] = s;
    __syncthreads();
    for (int off = 128; off > 0; off >>= 1) {
        if (tid < off) red[tid] += red[tid + off];
        __syncthreads();
    }
    if (tid == 0) {
        out_scalars[0] = loss[0] * (1.0f / ((float)M_TOTAL * (float)CODE_DIM)); // commit_loss
        out_scalars[1] = expf(-red[0]);                                          // perplexity
    }
}

extern "C" void kernel_launch(void* const* d_in, const int* in_sizes, int n_in,
                              void* d_out, int out_size, void* d_ws, size_t ws_size,
                              hipStream_t stream) {
    const float* x  = (const float*)d_in[0];   // [32, 512, 2048] f32
    const float* cb = (const float*)d_in[1];   // [512, 512] f32
    float* out = (float*)d_out;

    // workspace layout
    char* ws = (char*)d_ws;
    unsigned int* bhi    = (unsigned int*)(ws);                       // 512 KB
    unsigned int* blo    = (unsigned int*)(ws + (512u << 10));        // 512 KB
    float*        c_sq   = (float*)(ws + (1024u << 10));              // 2 KB
    float*        counts = (float*)(ws + (1024u << 10) + 4096);       // 2 KB
    float*        loss   = (float*)(ws + (1024u << 10) + 8192);       // 4 B

    // d_out layout: x_out [N*D*T] | commit_loss | perplexity | code_idx [M] (as float)
    float* out_x       = out;
    float* out_scalars = out + (size_t)NBATCH * CODE_DIM * TLEN;
    float* out_idx     = out_scalars + 2;

    vq_zero        <<<2,   256, 0, stream>>>(counts, loss);
    vq_prep_csq    <<<NB_CODE, 256, 0, stream>>>(cb, c_sq);
    vq_prep_swizzle<<<KTILES * DCHUNKS, 256, 0, stream>>>(cb, bhi, blo);
    vq_main        <<<M_TOTAL / 16 / 8, 256, 0, stream>>>(x, cb, bhi, blo, c_sq,
                                                          out_x, out_idx, counts, loss);
    vq_final       <<<1,   256, 0, stream>>>(counts, loss, out_scalars);
}
